// SpectralConv2d_71270687310570
// MI455X (gfx1250) — matrix operand order repaired and checked by reading, not compiled, not run on hardware
//
#include <hip/hip_runtime.h>
#include <hip/hip_bf16.h>

// ---------------------------------------------------------------------------
// FNO SpectralConv2d for MI455X (gfx1250, wave32, WMMA).
//
// x:[16,32,256,256] f32; w1,w2:[32,32,16,16] complex64 (interleaved f32).
// out = irfft2( scatter( einsum(truncated rfft2(x), w) ) )  -> [16,32,256,256]
//
//   S1: W-DFT (keep kw=0..15): GEMM M=131072 K=256 N=32 -> v_wmma_f32_16x16x32_f16
//       (twiddle B fragments computed once per block into LDS fragment layout;
//        read back as 2x ds_load_b128 per WMMA operand -- no spills, no global)
//   S2: H-DFT onto 32 kept kh rows                      -> f32 VALU + cos LUT
//   S3: per-mode complex channel mix [16x32x32]         -> v_wmma_f32_16x16x4_f32
//   S4: inverse H-DFT over 32 kept rows                 -> f32 VALU + cos LUT
//   S5: inverse rfft along W: GEMM M=131072 K=32 N=256  -> v_wmma_f32_16x16x32_f16
//       (16 N-tile B fragments in LDS; scale 1/(H*W) + Hermitian x2 folded in)
// ---------------------------------------------------------------------------

typedef __attribute__((ext_vector_type(16))) _Float16 v16h;
typedef __attribute__((ext_vector_type(8)))  float    v8f;
typedef __attribute__((ext_vector_type(2)))  float    v2f;

#define BATCH 16
#define CIN   32
#define COUT  32
#define HH    256
#define WW    256
#define M1    16
#define M2    16
#define TWO_PI 6.28318530717958647692f

// ---------------------------------------------------------------------------
// Kernel 1: forward W-DFT.  One wave per block, 8 row-tiles per wave
// (1024 blocks x 32 thr).  B fragments (cos and -sin) for all 8 K-steps are
// computed once per block into LDS in *fragment layout*: lane's 16 halves
// contiguous (32B) -> each WMMA B operand is two ds_load_b128.
// Fragment half hv of lane maps to K index k = ks*32 + (lane>>4)*16 + hv.
// A (x rows, f32->f16) staged through LDS per tile/K-step and read in the same
// contiguous K order as B (k = (lane>>4)*16 + s).  X1 layout: [bc*256+h][kw] (re and im planes).
// ---------------------------------------------------------------------------
__global__ void fno_dft_w(const float* __restrict__ x,
                          float* __restrict__ x1re,
                          float* __restrict__ x1im) {
  __shared__ __attribute__((aligned(32))) _Float16 atile[16 * 32];      // 1 KB
  __shared__ __attribute__((aligned(32))) _Float16 lbc[8 * 32 * 16];    // 8 KB
  __shared__ __attribute__((aligned(32))) _Float16 lbs[8 * 32 * 16];    // 8 KB
  const int lane = threadIdx.x;            // 0..31
  const int m    = lane & 15;              // A row / D column group
  const int kg   = lane >> 4;              // lane-half (K sub-block)
  const int n    = lane & 15;              // B column (kw)

  // ---- build B fragments in LDS (once per block) ----
  #pragma unroll
  for (int ks = 0; ks < 8; ++ks) {
    _Float16* pc = &lbc[(ks * 32 + lane) * 16];
    _Float16* ps = &lbs[(ks * 32 + lane) * 16];
    #pragma unroll
    for (int hv = 0; hv < 16; ++hv) {
      int   w  = ks * 32 + kg * 16 + hv;               // K index = w
      float th = TWO_PI * (float)((w * n) & 255) * (1.0f / 256.0f);
      pc[hv] = (_Float16)__cosf(th);
      ps[hv] = (_Float16)(-__sinf(th));
    }
  }
  __syncthreads();

  for (int t = 0; t < 8; ++t) {
    const long rowbase = ((long)blockIdx.x * 8 + t) * 16;  // row = bc*256+h
    v8f accR = {};
    v8f accI = {};
    if (t < 7) __builtin_prefetch(x + (rowbase + 16 + (lane & 15)) * WW, 0, 1);

    for (int ks = 0; ks < 8; ++ks) {
      const int kb = ks * 32;
      { // cooperative stage of A tile [16 rows x 32 w] -> f16 LDS
        const int row = lane & 15, seg = lane >> 4;
        const float* src = x + (rowbase + row) * WW + kb + seg * 16;
        #pragma unroll
        for (int i = 0; i < 16; i += 4) {
          float4 v = *(const float4*)(src + i);
          _Float16* d = &atile[row * 32 + seg * 16 + i];
          d[0] = (_Float16)v.x; d[1] = (_Float16)v.y;
          d[2] = (_Float16)v.z; d[3] = (_Float16)v.w;
        }
      }
      __syncthreads();
      // A fragment: lanes0-15 row m hold K 0..15; lanes16-31 hold K 16..31
      v16h a;
      #pragma unroll
      for (int j = 0; j < 8; ++j) {
        int pk = kg * 16 + 2 * j;
        a[2 * j]     = atile[m * 32 + pk];
        a[2 * j + 1] = atile[m * 32 + pk + 1];
      }
      v16h bc = *(const v16h*)&lbc[(ks * 32 + lane) * 16];
      v16h bs = *(const v16h*)&lbs[(ks * 32 + lane) * 16];
      accR = __builtin_amdgcn_wmma_f32_16x16x32_f16(false, a, false, bc,
                                                    (short)0, accR, false, false);
      accI = __builtin_amdgcn_wmma_f32_16x16x32_f16(false, a, false, bs,
                                                    (short)0, accI, false, false);
      __syncthreads();
    }

    // D layout: VGPR v -> (M=v or v+8, N=lane&15)
    const int col = lane & 15, mo = kg * 8;
    #pragma unroll
    for (int v = 0; v < 8; ++v) {
      long r = rowbase + v + mo;
      x1re[r * 16 + col] = accR[v];
      x1im[r * 16 + col] = accI[v];
    }
  }
}

// ---------------------------------------------------------------------------
// Kernel 2: H-DFT onto the 32 kept frequencies.  One block per (b,cin);
// 512 threads = (j in 0..31) x (kw in 0..15).  cos LUT in LDS:
// theta = 2pi*(h*kh mod 256)/256, sin t = cos(t + 3pi/2).
// X2 layout: [(b*32+cin)][j*16+kw].
// ---------------------------------------------------------------------------
__global__ void fno_dft_h(const float* __restrict__ x1re,
                          const float* __restrict__ x1im,
                          float* __restrict__ x2re,
                          float* __restrict__ x2im) {
  __shared__ float sre[HH * 16];
  __shared__ float sim[HH * 16];
  __shared__ float ctab[256];
  const int tid = threadIdx.x;     // 0..511
  const int bc  = blockIdx.x;      // 0..511
  if (tid < 256) ctab[tid] = __cosf(TWO_PI * (1.0f / 256.0f) * (float)tid);
  const float* gre = x1re + (long)bc * (HH * 16);
  const float* gim = x1im + (long)bc * (HH * 16);
  for (int i = tid; i < HH * 16; i += 512) { sre[i] = gre[i]; sim[i] = gim[i]; }
  __syncthreads();

  const int j  = tid >> 4, kw = tid & 15;
  const int kh = (j < 16) ? j : (224 + j);     // 240 + (j-16)
  float ar = 0.0f, ai = 0.0f;
  for (int h = 0; h < HH; ++h) {
    float a = sre[h * 16 + kw], b = sim[h * 16 + kw];
    int   t = (h * kh) & 255;
    float c = ctab[t];
    float s = ctab[(t + 192) & 255];           // = sin(theta)
    ar += a * c + b * s;                       // e^{-i theta} * (a + ib)
    ai += b * c - a * s;
  }
  x2re[(long)bc * 512 + tid] = ar;
  x2im[(long)bc * 512 + tid] = ai;
}

// ---------------------------------------------------------------------------
// Kernel 3: complex channel mixing per mode with v_wmma_f32_16x16x4_f32.
// One wave per (mode, cout-half).  M=batch(16), K=cin(32, 8 steps of 4),
// N=cout(16).  Complex = 4 real accumulators; Re = rr-ii, Im = ri+ir (VALU).
// f32 A 16x4: VGPR0 K=kg*2, VGPR1 K=kg*2+1 (kg = lane>>4); B mirrors A.
// X3 layout: [(b*32+cout)][j*16+kw].
// ---------------------------------------------------------------------------
__global__ void fno_mix(const float* __restrict__ x2re,
                        const float* __restrict__ x2im,
                        const float* __restrict__ w1,
                        const float* __restrict__ w2,
                        float* __restrict__ x3re,
                        float* __restrict__ x3im) {
  const int lane  = threadIdx.x;
  const int mode  = blockIdx.x;        // j*16 + kw
  const int ntile = blockIdx.y;        // 0..1
  const int j  = mode >> 4, kw = mode & 15, jm = j & 15;
  const float* wsel = (j < 16) ? w1 : w2;
  const int m     = lane & 15;         // batch row for A; column for B/D
  const int kg    = lane >> 4;
  const int coutN = ntile * 16 + m;

  v8f crr = {}, cii = {}, cri = {}, cir = {};
  #pragma unroll
  for (int ks = 0; ks < 8; ++ks) {
    const int k0 = ks * 4 + kg * 2;    // cin for VGPR0; VGPR1 = k0+1
    long ia0 = (long)(m * CIN + k0) * 512 + mode;
    long ia1 = (long)(m * CIN + k0 + 1) * 512 + mode;
    v2f arR = { x2re[ia0], x2re[ia1] };
    v2f arI = { x2im[ia0], x2im[ia1] };
    long ib0 = 2 * ((((long)k0 * COUT + coutN) * M1 + jm) * M2 + kw);
    long ib1 = 2 * ((((long)(k0 + 1) * COUT + coutN) * M1 + jm) * M2 + kw);
    v2f brR = { wsel[ib0],     wsel[ib1]     };
    v2f brI = { wsel[ib0 + 1], wsel[ib1 + 1] };
    crr = __builtin_amdgcn_wmma_f32_16x16x4_f32(false, arR, false, brR,
                                                (short)0, crr, false, false);
    cii = __builtin_amdgcn_wmma_f32_16x16x4_f32(false, arI, false, brI,
                                                (short)0, cii, false, false);
    cri = __builtin_amdgcn_wmma_f32_16x16x4_f32(false, arR, false, brI,
                                                (short)0, cri, false, false);
    cir = __builtin_amdgcn_wmma_f32_16x16x4_f32(false, arI, false, brR,
                                                (short)0, cir, false, false);
  }
  v8f re = crr - cii;
  v8f im = cri + cir;
  const int mo = kg * 8;
  #pragma unroll
  for (int v = 0; v < 8; ++v) {
    int  b   = v + mo;                 // D row = batch index
    long idx = (long)(b * COUT + coutN) * 512 + mode;
    x3re[idx] = re[v];
    x3im[idx] = im[v];
  }
}

// ---------------------------------------------------------------------------
// Kernel 4: inverse H-DFT (only 32 nonzero rows).  One block per (b,cout);
// 256 threads = h.  e^{+i theta}.  Output Y1 interleaved re/im per kw so the
// inverse-W GEMM sees K=32 contiguous: y1[(bo*256+h)*32 + 2kw + {0,1}].
// ---------------------------------------------------------------------------
__global__ void fno_idft_h(const float* __restrict__ x3re,
                           const float* __restrict__ x3im,
                           float* __restrict__ y1) {
  __shared__ float sre[512];
  __shared__ float sim[512];
  __shared__ float ctab[256];
  const int tid = threadIdx.x;     // 0..255 == h
  const int bo  = blockIdx.x;      // 0..511
  ctab[tid] = __cosf(TWO_PI * (1.0f / 256.0f) * (float)tid);
  sre[tid]       = x3re[(long)bo * 512 + tid];
  sre[tid + 256] = x3re[(long)bo * 512 + tid + 256];
  sim[tid]       = x3im[(long)bo * 512 + tid];
  sim[tid + 256] = x3im[(long)bo * 512 + tid + 256];
  __syncthreads();

  const int h = tid;
  float outv[32];
  for (int kw = 0; kw < 16; ++kw) {
    float ar = 0.0f, ai = 0.0f;
    #pragma unroll
    for (int j = 0; j < 32; ++j) {
      int   kh = (j < 16) ? j : (224 + j);
      int   t  = (h * kh) & 255;
      float c  = ctab[t];
      float s  = ctab[(t + 192) & 255];          // sin(theta)
      float a  = sre[j * 16 + kw], b = sim[j * 16 + kw];
      ar += a * c - b * s;                       // e^{+i theta} * (a + ib)
      ai += a * s + b * c;
    }
    outv[2 * kw]     = ar;
    outv[2 * kw + 1] = ai;
  }
  float* dst = y1 + ((long)bo * HH + h) * 32;
  #pragma unroll
  for (int i = 0; i < 32; ++i) dst[i] = outv[i];
}

// ---------------------------------------------------------------------------
// Kernel 5: inverse W transform (real output).  One wave per block, 8
// row-tiles per wave (1024 blocks).  The 16 N-tiles' B fragments are built
// once per block into LDS fragment layout (16 KB); K=32 = one WMMA step.
// Fragment half hv maps to K index k = (lane>>4)*16 + hv; k = 2*kw + ri
// with ri=0 -> g*cos, ri=1 -> -g*sin, g = (kw?2:1)/65536 (irfft scale +
// Hermitian doubling).  out[row*256 + w], row = (b*32+cout)*256 + h.
// ---------------------------------------------------------------------------
__global__ void fno_idft_w(const float* __restrict__ y1,
                           float* __restrict__ out) {
  __shared__ __attribute__((aligned(32))) _Float16 atile[16 * 32];      // 1 KB
  __shared__ __attribute__((aligned(32))) _Float16 lbt[16 * 32 * 16];   // 16 KB
  const int lane = threadIdx.x;
  const int m = lane & 15, kg = lane >> 4, n = lane & 15;

  // ---- build all 16 N-tiles' B fragments in LDS (once per block) ----
  #pragma unroll
  for (int nt = 0; nt < 16; ++nt) {
    _Float16* p = &lbt[(nt * 32 + lane) * 16];
    const int w = nt * 16 + n;
    #pragma unroll
    for (int hv = 0; hv < 16; ++hv) {
      int   k  = kg * 16 + hv;                    // 0..31 = 2*kw + ri
      int   kw = k >> 1, ri = k & 1;
      float g  = (kw == 0 ? 1.0f : 2.0f) * (1.0f / 65536.0f);
      float th = TWO_PI * (float)((w * kw) & 255) * (1.0f / 256.0f);
      p[hv] = (_Float16)(ri == 0 ? g * __cosf(th) : -g * __sinf(th));
    }
  }
  __syncthreads();

  for (int t = 0; t < 8; ++t) {
    const long rowbase = ((long)blockIdx.x * 8 + t) * 16;
    if (t < 7) __builtin_prefetch(y1 + (rowbase + 16 + (lane & 15)) * 32, 0, 1);
    { // stage A tile [16 rows x 32 k] f32 -> f16 LDS
      const int row = lane & 15, seg = lane >> 4;
      const float* src = y1 + (rowbase + row) * 32 + seg * 16;
      #pragma unroll
      for (int i = 0; i < 16; i += 4) {
        float4 v = *(const float4*)(src + i);
        _Float16* d = &atile[row * 32 + seg * 16 + i];
        d[0] = (_Float16)v.x; d[1] = (_Float16)v.y;
        d[2] = (_Float16)v.z; d[3] = (_Float16)v.w;
      }
    }
    __syncthreads();
    v16h a;
    #pragma unroll
    for (int j = 0; j < 8; ++j) {
      int pk = kg * 16 + 2 * j;
      a[2 * j]     = atile[m * 32 + pk];
      a[2 * j + 1] = atile[m * 32 + pk + 1];
    }

    const int mo = kg * 8;
    #pragma unroll
    for (int nt = 0; nt < 16; ++nt) {
      v16h b = *(const v16h*)&lbt[(nt * 32 + lane) * 16];
      v8f acc = {};
      acc = __builtin_amdgcn_wmma_f32_16x16x32_f16(false, a, false, b,
                                                   (short)0, acc, false, false);
      #pragma unroll
      for (int v = 0; v < 8; ++v)
        out[(rowbase + v + mo) * WW + nt * 16 + n] = acc[v];
    }
    __syncthreads();
  }
}

// ---------------------------------------------------------------------------
// Host launcher.  Scratch layout in d_ws (~37.7 MB, all buffers fully
// written before read; no cross-call state).
// ---------------------------------------------------------------------------
extern "C" void kernel_launch(void* const* d_in, const int* in_sizes, int n_in,
                              void* d_out, int out_size, void* d_ws, size_t ws_size,
                              hipStream_t stream) {
  const float* x  = (const float*)d_in[0];
  const float* w1 = (const float*)d_in[1];   // complex64 interleaved
  const float* w2 = (const float*)d_in[2];

  char* ws = (char*)d_ws;
  const size_t NX1 = (size_t)BATCH * CIN  * HH * M2;       // 2,097,152
  const size_t NX2 = (size_t)BATCH * CIN  * 2 * M1 * M2;   //   262,144
  const size_t NY1 = (size_t)BATCH * COUT * HH * 2 * M2;   // 4,194,304

  size_t off = 0;
  float* x1re = (float*)(ws + off); off += NX1 * 4;
  float* x1im = (float*)(ws + off); off += NX1 * 4;
  float* x2re = (float*)(ws + off); off += NX2 * 4;
  float* x2im = (float*)(ws + off); off += NX2 * 4;
  float* x3re = (float*)(ws + off); off += NX2 * 4;
  float* x3im = (float*)(ws + off); off += NX2 * 4;
  float* y1   = (float*)(ws + off); off += NY1 * 4;

  float* out = (float*)d_out;

  // S1: 8192 row-tiles, 8 per wave-block
  fno_dft_w<<<1024, 32, 0, stream>>>(x, x1re, x1im);

  // S2: one block per (b,cin)
  fno_dft_h<<<BATCH * CIN, 512, 0, stream>>>(x1re, x1im, x2re, x2im);

  // S3: one wave per (mode, cout-half)
  fno_mix<<<dim3(2 * M1 * M2, 2), 32, 0, stream>>>(x2re, x2im, w1, w2,
                                                   x3re, x3im);

  // S4: one block per (b,cout)
  fno_idft_h<<<BATCH * COUT, 256, 0, stream>>>(x3re, x3im, y1);

  // S5: 8192 row-tiles, 8 per wave-block, writes all of d_out
  fno_idft_w<<<1024, 32, 0, stream>>>(y1, out);
}
